// GraphConvNet_65257733095607
// MI455X (gfx1250) — matrix operand order repaired and checked by reading, not compiled, not run on hardware
//
#include <hip/hip_runtime.h>
#include <math.h>

// ---------------------------------------------------------------------------
// CDNA5 (gfx1250) wave32 WMMA types
// ---------------------------------------------------------------------------
typedef __attribute__((ext_vector_type(16))) __bf16 v16bf;
typedef __attribute__((ext_vector_type(8)))  float  v8f;
typedef __attribute__((ext_vector_type(4)))  int    v4i;

union Frag16 {
    v16bf    v;
    unsigned u[8];
    uint4    q[2];
};

__device__ __forceinline__ unsigned short f2bf(float f) {
    unsigned u = __builtin_bit_cast(unsigned, f);
    unsigned r = u + 0x7FFFu + ((u >> 16) & 1u);   // round-to-nearest-even
    return (unsigned short)(r >> 16);
}

// ---------------------------------------------------------------------------
// Weight prep: fused GraphConv weight  Wb[n*256+k] = (k<128 ? wrel[n][k]
//                                                          : wroot[n][k-128])
// ---------------------------------------------------------------------------
__global__ void prep_conv_w(const float* __restrict__ wrel,
                            const float* __restrict__ wroot,
                            unsigned short* __restrict__ Wb) {
    int idx = blockIdx.x * 256 + threadIdx.x;      // 0..32767
    int n = idx >> 8, k = idx & 255;
    float v = (k < 128) ? wrel[n * 128 + k] : wroot[n * 128 + (k - 128)];
    Wb[idx] = f2bf(v);
}

// Generic linear weight: Wb[n*K + k] = w[n*K + k]
__global__ void prep_lin_w(const float* __restrict__ w,
                           unsigned short* __restrict__ Wb, int N, int K) {
    int idx = blockIdx.x * 256 + threadIdx.x;
    if (idx >= N * K) return;
    int n = idx / K, k = idx % K;
    Wb[idx] = f2bf(w[n * K + k]);
}

// ---------------------------------------------------------------------------
// Edge scatter: agg[dst] += h[src] (128 floats/row), one wave per edge,
// lane handles 4 consecutive floats. -1 src/dst == masked edge.
// agg is L2-resident (<=32MB vs 192MB L2) so f32 atomics resolve in L2.
// ---------------------------------------------------------------------------
__global__ __launch_bounds__(256) void scatter_add(
        const float* __restrict__ h, const int* __restrict__ src,
        const int* __restrict__ dst, float* __restrict__ agg, int E) {
    int e = blockIdx.x * 8 + (threadIdx.x >> 5);
    if (e >= E) return;
    int s = src[e], d = dst[e];
    if (s < 0 || d < 0) return;
    int c = (threadIdx.x & 31) * 4;
    const float4 v = *(const float4*)(h + (size_t)s * 128 + c);
    float* o = agg + (size_t)d * 128 + c;
    atomicAdd(o + 0, v.x);
    atomicAdd(o + 1, v.y);
    atomicAdd(o + 2, v.z);
    atomicAdd(o + 3, v.w);
}

// ---------------------------------------------------------------------------
// Fused GraphConv GEMM:
//   out[m, 0:128] = relu( [agg(m,0:128) | h(m,0:128)] @ Wb(256x128) + bias )
// Block = 256 threads = 8 waves, 64 output rows per block.
// Each wave owns one 16-col tile and 4 M-subtiles (reuses each B fragment 4x).
// B (64KB bf16) is async-DMA'd to LDS while VALU converts the A panel.
// Fragments are read from LDS as 16B chunks -> ds_load_b128.
// ---------------------------------------------------------------------------
__global__ __launch_bounds__(256) void conv_gemm(
        const float* __restrict__ agg, const float* __restrict__ h,
        const unsigned short* __restrict__ Wb, const float* __restrict__ bias,
        float* __restrict__ out) {
    __shared__ unsigned short As[64 * 256];    // 32 KB bf16 A panel (LDS base)
    __shared__ unsigned short Bs[256 * 128];   // 64 KB bf16 weights
    const int m0 = blockIdx.x * 64;
    const int t  = threadIdx.x;

    // ---- stage B via CDNA5 async global->LDS DMA (ASYNCcnt-tracked)
#if __has_builtin(__builtin_amdgcn_global_load_async_to_lds_b128)
    {
        typedef __attribute__((address_space(1))) v4i* gptr_t;
        typedef __attribute__((address_space(3))) v4i* lptr_t;
        gptr_t gsrc = (gptr_t)Wb;
        lptr_t ldst = (lptr_t)Bs;
#pragma unroll
        for (int j = 0; j < 16; ++j) {
            int idx = t + j * 256;             // 4096 x 16B chunks
            __builtin_amdgcn_global_load_async_to_lds_b128(gsrc + idx, ldst + idx, 0, 0);
        }
    }
#else
    {
        const uint4* srcp = (const uint4*)Wb;
        uint4*       dstp = (uint4*)Bs;
#pragma unroll
        for (int j = 0; j < 16; ++j) dstp[t + j * 256] = srcp[t + j * 256];
    }
#endif

    // ---- stage A = [agg | h] rows m0..m0+63, fp32 -> bf16 (overlaps async B)
    {
        int r  = t >> 2;
        int c0 = (t & 3) << 6;
        const float* rowA = agg + (size_t)(m0 + r) * 128;
        const float* rowH = h   + (size_t)(m0 + r) * 128;
#pragma unroll 8
        for (int j = 0; j < 64; ++j) {
            int c = c0 + j;
            float v = (c < 128) ? rowA[c] : rowH[c - 128];
            As[r * 256 + c] = f2bf(v);
        }
    }
#if __has_builtin(__builtin_amdgcn_global_load_async_to_lds_b128)
#if __has_builtin(__builtin_amdgcn_s_wait_asynccnt)
    __builtin_amdgcn_s_wait_asynccnt(0);
#else
    asm volatile("s_wait_asynccnt 0x0" ::: "memory");
#endif
#endif
    __syncthreads();

    const int wv   = t >> 5;
    const int lane = t & 31;
    const int n0   = wv << 4;
    const int mrow = lane & 15;
    const int half = lane >> 4;
    const uint4* As128 = (const uint4*)As;     // 16B = 8 bf16 chunks
    const uint4* Bs128 = (const uint4*)Bs;

    v8f acc[4] = {v8f{}, v8f{}, v8f{}, v8f{}};
#pragma unroll
    for (int kb = 0; kb < 8; ++kb) {           // K = 256 in steps of 32
        const int k0 = kb * 32;
        Frag16 bu;                             // lane L holds B col n0+L%16, K runs k0+8h.., k0+16+8h..
        {
            int bbase = (((n0 + mrow) * 256 + k0 + (half << 3)) >> 3);
            bu.q[0] = Bs128[bbase];
            bu.q[1] = Bs128[bbase + 2];
        }
#pragma unroll
        for (int mt = 0; mt < 4; ++mt) {
            Frag16 au;                         // ISA 16-bit A layout: two 8-elem runs
            int abase = (((mt * 16 + mrow) * 256 + k0 + (half << 3)) >> 3);
            au.q[0] = As128[abase];            // K = k0+8h .. +7
            au.q[1] = As128[abase + 2];        // K = k0+16+8h .. +7
            acc[mt] = __builtin_amdgcn_wmma_f32_16x16x32_bf16(
                    false, au.v, false, bu.v, (short)0, acc[mt], false, false);
        }
    }

    const int   nn = n0 + (lane & 15);
    const float bz = bias[nn];
#pragma unroll
    for (int mt = 0; mt < 4; ++mt) {
#pragma unroll
        for (int r = 0; r < 8; ++r) {          // C/D layout: M=r (+8 hi half)
            int   mm  = m0 + mt * 16 + r + (half << 3);
            float val = acc[mt][r] + bz;
            out[(size_t)mm * 128 + nn] = val > 0.f ? val : 0.f;
        }
    }
}

// ---------------------------------------------------------------------------
// Generic WMMA GEMM with bias + relu:  out(MxN) = relu(A(MxK) @ Wb(KxN) + b)
// blockDim = 32*(N/16); grid = M/16. A is fp32, converted inline.
// ---------------------------------------------------------------------------
__global__ __launch_bounds__(256) void lin_gemm(
        const float* __restrict__ A, const unsigned short* __restrict__ Wb,
        const float* __restrict__ bias, float* __restrict__ out,
        int K, int N) {
    const int m0   = blockIdx.x * 16;
    const int t    = threadIdx.x;
    const int wv   = t >> 5, lane = t & 31;
    const int n0   = wv << 4;
    const int mrow = lane & 15, half = lane >> 4;
    const uint4* W128 = (const uint4*)Wb;
    const float* arow = A + (size_t)(m0 + mrow) * K;

    v8f acc = {};
    for (int k0 = 0; k0 < K; k0 += 32) {
        Frag16 au, bu;
#pragma unroll
        for (int v = 0; v < 8; ++v) {
            int kk = k0 + ((v & 4) ? 16 : 0) + (half << 3) + ((v & 3) << 1);
            unsigned lo = f2bf(arow[kk]);
            unsigned hi = f2bf(arow[kk + 1]);
            au.u[v] = lo | (hi << 16);
        }
        int bbase = (((n0 + mrow) * K + k0 + (half << 3)) >> 3);
        bu.q[0] = W128[bbase];
        bu.q[1] = W128[bbase + 2];
        acc = __builtin_amdgcn_wmma_f32_16x16x32_bf16(
                false, au.v, false, bu.v, (short)0, acc, false, false);
    }

    const int   nn = n0 + (lane & 15);
    const float bz = bias[nn];
#pragma unroll
    for (int r = 0; r < 8; ++r) {
        int   mm  = m0 + r + (half << 3);
        float val = acc[r] + bz;
        out[(size_t)mm * N + nn] = val > 0.f ? val : 0.f;
    }
}

// ---------------------------------------------------------------------------
// TopK pooling: one block (128 thr) per graph. Stable O(n^2) rank selection
// matching jax.lax.top_k tie-breaking (lower index wins).
// ---------------------------------------------------------------------------
__global__ __launch_bounds__(128) void topk_pool_k(
        const float* __restrict__ h, const float* __restrict__ w,
        int n_nodes, int k, float* __restrict__ vals, int* __restrict__ perm,
        int* __restrict__ new_id) {
    __shared__ float sc[128];
    const int b = blockIdx.x;
    const int t = threadIdx.x;

    float nw = 0.f;
    for (int c = 0; c < 128; ++c) nw += w[c] * w[c];
    const float inv = 1.0f / sqrtf(nw);

    if (t < n_nodes) {
        const float* row = h + ((size_t)b * n_nodes + t) * 128;
        float dot = 0.f;
        for (int c = 0; c < 128; ++c) dot += row[c] * w[c];
        sc[t] = tanhf(dot * inv);
    }
    __syncthreads();

    if (t < n_nodes) {
        float my = sc[t];
        int r = 0;
        for (int j = 0; j < n_nodes; ++j) {
            float o = sc[j];
            if (o > my || (o == my && j < t)) ++r;
        }
        int node = b * n_nodes + t;
        if (r < k) {
            perm[b * k + r]   = node;
            vals[b * k + r]   = my;
            new_id[node]      = b * k + r;
        } else {
            new_id[node] = -1;
        }
    }
}

// hout[i] = h[perm[i]] * vals[i]   (one wave per row, float4 per lane)
__global__ __launch_bounds__(256) void gather_scale(
        const float* __restrict__ h, const int* __restrict__ perm,
        const float* __restrict__ vals, float* __restrict__ hout, int Mout) {
    int i = blockIdx.x * 8 + (threadIdx.x >> 5);
    if (i >= Mout) return;
    int c = (threadIdx.x & 31) * 4;
    float s = vals[i];
    const float4 v = *(const float4*)(h + (size_t)perm[i] * 128 + c);
    float4 o = make_float4(v.x * s, v.y * s, v.z * s, v.w * s);
    *(float4*)(hout + (size_t)i * 128 + c) = o;
}

// Remap edges through new_id; -1 == dropped edge.
__global__ void remap_edges(const int* __restrict__ si, const int* __restrict__ di,
                            const int* __restrict__ new_id,
                            int* __restrict__ so, int* __restrict__ dd, int E) {
    int e = blockIdx.x * blockDim.x + threadIdx.x;
    if (e >= E) return;
    int s = si[e], d = di[e];
    int ns = -1, nd = -1;
    if (s >= 0 && d >= 0) { ns = new_id[s]; nd = new_id[d]; }
    bool ok = (ns >= 0 && nd >= 0);
    so[e] = ok ? ns : -1;
    dd[e] = ok ? nd : -1;
}

// z[b, 0:128] += max over k rows, z[b, 128:256] += mean over k rows
__global__ __launch_bounds__(128) void readout_add(
        const float* __restrict__ hp, int k, float* __restrict__ z) {
    int b = blockIdx.x, c = threadIdx.x;
    const float* base = hp + (size_t)b * k * 128 + c;
    float mx = -3.0e38f, sm = 0.f;
    for (int i = 0; i < k; ++i) {
        float v = base[(size_t)i * 128];
        mx = fmaxf(mx, v);
        sm += v;
    }
    z[b * 256 + c]       += mx;
    z[b * 256 + 128 + c] += sm / (float)k;
}

// ---------------------------------------------------------------------------
// Launch
// ---------------------------------------------------------------------------
extern "C" void kernel_launch(void* const* d_in, const int* in_sizes, int n_in_,
                              void* d_out, int out_size, void* d_ws, size_t ws_size,
                              hipStream_t stream) {
    (void)n_in_; (void)out_size; (void)ws_size;
    const float* x        = (const float*)d_in[0];
    const int*   edge     = (const int*)d_in[1];
    const float* c1_wrel  = (const float*)d_in[2];
    const float* c1_brel  = (const float*)d_in[3];
    const float* c1_wroot = (const float*)d_in[4];
    const float* p1_w     = (const float*)d_in[5];
    const float* c2_wrel  = (const float*)d_in[6];
    const float* c2_brel  = (const float*)d_in[7];
    const float* c2_wroot = (const float*)d_in[8];
    const float* p2_w     = (const float*)d_in[9];
    const float* c3_wrel  = (const float*)d_in[10];
    const float* c3_brel  = (const float*)d_in[11];
    const float* c3_wroot = (const float*)d_in[12];
    const float* p3_w     = (const float*)d_in[13];
    const float* lin1_w   = (const float*)d_in[14];
    const float* lin1_b   = (const float*)d_in[15];
    const float* lin2_w   = (const float*)d_in[16];
    const float* lin2_b   = (const float*)d_in[17];
    float* out = (float*)d_out;

    const int E    = in_sizes[1] / 2;              // 524288
    const int* es0 = edge;
    const int* ed0 = edge + E;

    // workspace carve-out
    char* p = (char*)d_ws;
    auto take = [&](size_t bytes) -> char* {
        char* q = p; p += (bytes + 255) & ~(size_t)255; return q;
    };
    float* agg   = (float*)take(65536ull * 128 * 4);
    float* hconv = (float*)take(65536ull * 128 * 4);
    float* hp    = (float*)take(32768ull * 128 * 4);
    float* z     = (float*)take(512ull * 256 * 4);
    float* t1    = (float*)take(512ull * 128 * 4);
    float* vals  = (float*)take(32768ull * 4);
    int*   perm  = (int*)take(32768ull * 4);
    int*   nid   = (int*)take(65536ull * 4);
    int*   eAs   = (int*)take((size_t)E * 4);
    int*   eAd   = (int*)take((size_t)E * 4);
    int*   eBs   = (int*)take((size_t)E * 4);
    int*   eBd   = (int*)take((size_t)E * 4);
    unsigned short* Wc1 = (unsigned short*)take(256 * 128 * 2);
    unsigned short* Wc2 = (unsigned short*)take(256 * 128 * 2);
    unsigned short* Wc3 = (unsigned short*)take(256 * 128 * 2);
    unsigned short* Wl1 = (unsigned short*)take(256 * 128 * 2);
    unsigned short* Wl2 = (unsigned short*)take(128 * 64 * 2);

    // weight prep (bf16, stored N x K, k contiguous)
    prep_conv_w<<<128, 256, 0, stream>>>(c1_wrel, c1_wroot, Wc1);
    prep_conv_w<<<128, 256, 0, stream>>>(c2_wrel, c2_wroot, Wc2);
    prep_conv_w<<<128, 256, 0, stream>>>(c3_wrel, c3_wroot, Wc3);
    prep_lin_w<<<128, 256, 0, stream>>>(lin1_w, Wl1, 128, 256);
    prep_lin_w<<<32, 256, 0, stream>>>(lin2_w, Wl2, 64, 128);

    (void)hipMemsetAsync(z, 0, 512 * 256 * 4, stream);

    // ---- stage 1: M=65536, pool 128 -> 64
    (void)hipMemsetAsync(agg, 0, 65536ull * 128 * 4, stream);
    scatter_add<<<E / 8, 256, 0, stream>>>(x, es0, ed0, agg, E);
    conv_gemm<<<65536 / 64, 256, 0, stream>>>(agg, x, Wc1, c1_brel, hconv);
    topk_pool_k<<<512, 128, 0, stream>>>(hconv, p1_w, 128, 64, vals, perm, nid);
    gather_scale<<<32768 / 8, 256, 0, stream>>>(hconv, perm, vals, hp, 32768);
    remap_edges<<<(E + 255) / 256, 256, 0, stream>>>(es0, ed0, nid, eAs, eAd, E);
    readout_add<<<512, 128, 0, stream>>>(hp, 64, z);

    // ---- stage 2: M=32768, pool 64 -> 32
    (void)hipMemsetAsync(agg, 0, 32768ull * 128 * 4, stream);
    scatter_add<<<E / 8, 256, 0, stream>>>(hp, eAs, eAd, agg, E);
    conv_gemm<<<32768 / 64, 256, 0, stream>>>(agg, hp, Wc2, c2_brel, hconv);
    topk_pool_k<<<512, 128, 0, stream>>>(hconv, p2_w, 64, 32, vals, perm, nid);
    gather_scale<<<16384 / 8, 256, 0, stream>>>(hconv, perm, vals, hp, 16384);
    remap_edges<<<(E + 255) / 256, 256, 0, stream>>>(eAs, eAd, nid, eBs, eBd, E);
    readout_add<<<512, 128, 0, stream>>>(hp, 32, z);

    // ---- stage 3: M=16384, pool 32 -> 16
    (void)hipMemsetAsync(agg, 0, 16384ull * 128 * 4, stream);
    scatter_add<<<E / 8, 256, 0, stream>>>(hp, eBs, eBd, agg, E);
    conv_gemm<<<16384 / 64, 256, 0, stream>>>(agg, hp, Wc3, c3_brel, hconv);
    topk_pool_k<<<512, 128, 0, stream>>>(hconv, p3_w, 32, 16, vals, perm, nid);
    gather_scale<<<8192 / 8, 256, 0, stream>>>(hconv, perm, vals, hp, 8192);
    readout_add<<<512, 128, 0, stream>>>(hp, 16, z);

    // ---- MLP head: (512x256)@(256x128) relu -> (512x128)@(128x64) relu
    lin_gemm<<<512 / 16, 256, 0, stream>>>(z, Wl1, lin1_b, t1, 256, 128);
    lin_gemm<<<512 / 16, 128, 0, stream>>>(t1, Wl2, lin2_b, out, 128, 64);
}
